// AttentionWithSelfAblation_48988396978480
// MI455X (gfx1250) — matrix operand order repaired and checked by reading, not compiled, not run on hardware
//
#include <hip/hip_runtime.h>

typedef __attribute__((ext_vector_type(16))) __bf16 v16bf;
typedef __attribute__((ext_vector_type(8)))  __bf16 v8bf;
typedef __attribute__((ext_vector_type(8)))  float  v8f;
typedef __attribute__((ext_vector_type(4)))  unsigned int u32x4;
typedef __attribute__((ext_vector_type(8)))  int i32x8;
typedef __attribute__((ext_vector_type(4)))  int i32x4;

#define BSZ  4
#define SSEQ 2048
#define HDIM 1024
#define NHEAD 16
#define HD   64
#define WIN  256

__device__ __forceinline__ unsigned short f2bf(float f) {
  unsigned int u = __builtin_bit_cast(unsigned int, f);
  u += 0x7FFFu + ((u >> 16) & 1u);          // round-to-nearest-even
  return (unsigned short)(u >> 16);
}

__device__ __forceinline__ v8f wmma_bf16(v16bf a, v16bf b, v8f c) {
  return __builtin_amdgcn_wmma_f32_16x16x32_bf16(
      /*neg_a=*/false, a, /*neg_b=*/false, b,
      /*c_mod=*/(short)0, c, /*reuse_a=*/false, /*reuse_b=*/false);
}

// Async global->LDS copy, 16B per lane, tracked by ASYNCcnt (CDNA5 path).
__device__ __forceinline__ void async_ld_b128(const unsigned short* g, void* lds) {
  unsigned loff = (unsigned)(uintptr_t)lds;                // LDS byte offset
  unsigned long long ga = (unsigned long long)(uintptr_t)g;
  asm volatile("global_load_async_to_lds_b128 %0, %1, off"
               :: "v"(loff), "v"(ga) : "memory");
}
__device__ __forceinline__ void wait_async0() {
  asm volatile("s_wait_asynccnt 0x0" ::: "memory");
}

// Tensor Data Mover: 2D tile DMA global->LDS, tracked by TENSORcnt.
// D# per CDNA5 ISA:
//  group0: [1:0]=count=1, [63:32]=lds_addr, [120:64]=global_addr, [127:126]=type=2
//  group1: [17:16]=data_size(1=2B), [79:48]=tensor_dim0, [111:80]=tensor_dim1,
//          [127:112]=tile_dim0, [143:128]=tile_dim1, [207:160]=tensor_dim0_stride
//  groups 2/3: zero (2D tile; dims >=3 unused)
__device__ __forceinline__ void tdm_load_2d(const void* gaddr, unsigned lds_off,
                                            unsigned tile_w, unsigned tile_h,
                                            unsigned tensor_w, unsigned tensor_h,
                                            unsigned row_stride_elems) {
  unsigned long long ga = (unsigned long long)(uintptr_t)gaddr;
  u32x4 g0;
  g0[0] = 1u;                                               // count=1 (user D#)
  g0[1] = lds_off;                                          // lds_addr bytes
  g0[2] = (unsigned)(ga & 0xFFFFFFFFu);                     // global_addr[31:0]
  g0[3] = (unsigned)((ga >> 32) & 0x1FFFFFFu) | (2u << 30); // ga[56:32] | type=2
  i32x8 g1;
  g1[0] = (int)(1u << 16);                                  // data_size = 2 bytes
  g1[1] = (int)((tensor_w & 0xFFFFu) << 16);                // tensor_dim0 lo
  g1[2] = (int)(((tensor_w >> 16) & 0xFFFFu) |
                ((tensor_h & 0xFFFFu) << 16));              // dim0 hi | dim1 lo
  g1[3] = (int)(((tensor_h >> 16) & 0xFFFFu) |
                ((tile_w & 0xFFFFu) << 16));                // dim1 hi | tile_dim0
  g1[4] = (int)(tile_h & 0xFFFFu);                          // tile_dim1 (tile_dim2=0)
  g1[5] = (int)row_stride_elems;                            // tensor_dim0_stride lo
  g1[6] = 0;
  g1[7] = 0;
  i32x4 z4 = {0, 0, 0, 0};
#if defined(__clang_major__) && __clang_major__ >= 23
  i32x8 z8 = {0, 0, 0, 0, 0, 0, 0, 0};
  __builtin_amdgcn_tensor_load_to_lds(g0, g1, z4, z4, z8, 0);
#else
  __builtin_amdgcn_tensor_load_to_lds(g0, g1, z4, z4, 0);
#endif
}

// A operand (16x32 bf16): lane holds row M = lane%16.
// lanes 0-15: elements = K 0..7 then 16..23 ; lanes 16-31: K 8..15 then 24..31
__device__ __forceinline__ v16bf load_fragA(const unsigned short* rowptr, int hi) {
  const int o = hi ? 8 : 0;
  v8bf lo  = *reinterpret_cast<const v8bf*>(rowptr + o);
  v8bf hi8 = *reinterpret_cast<const v8bf*>(rowptr + o + 16);
  v16bf r;
#pragma unroll
  for (int i = 0; i < 8; ++i) { r[i] = lo[i]; r[i + 8] = hi8[i]; }
  return r;
}

// B operand (32x16 bf16): lane holds column N = lane%16.
// lanes 0-15: K 0..7 then 16..23 ; lanes 16-31: K 8..15 then 24..31
__device__ __forceinline__ v16bf load_fragB(const unsigned short* colptr, int hi) {
  const int o = hi ? 8 : 0;
  v8bf lo  = *reinterpret_cast<const v8bf*>(colptr + o);
  v8bf hi8 = *reinterpret_cast<const v8bf*>(colptr + o + 16);
  v16bf r;
#pragma unroll
  for (int i = 0; i < 8; ++i) { r[i] = lo[i]; r[i + 8] = hi8[i]; }
  return r;
}

__global__ void cvt_bf16_kernel(const float* __restrict__ in,
                                unsigned short* __restrict__ out, int n) {
  int i = blockIdx.x * blockDim.x + threadIdx.x;
  if (i < n) out[i] = f2bf(in[i]);
}

// C = A[M,K] * W[N,K]^T.  Tile: 128(M) x 64(N) per 256-thread block, K-step 32.
// Double-buffered: A tile via TDM (wave 0, TENSORcnt), W tile via per-lane
// async global->LDS (ASYNCcnt); tile k+1 streams in while WMMAs consume tile k.
// mode 0: store bf16 row-major [M,NTOT]
// mode 1: store bf16 transposed per head: out[((b*16+h)*64+d)*Sdim + s]
// mode 2: store f32 [M,NTOT] + bias[n]
__global__ __launch_bounds__(256) void gemm_bf16_kernel(
    const unsigned short* __restrict__ A,
    const unsigned short* __restrict__ W,
    unsigned short* __restrict__ outbf,
    float* __restrict__ outf,
    const float* __restrict__ bias,
    int K, int NTOT, int Sdim, int mode) {
  __shared__ unsigned short As[2][128][32];   // 16 KB
  __shared__ unsigned short Ws[2][64][32];    //  8 KB
  const int m0 = blockIdx.y * 128;
  const int n0 = blockIdx.x * 64;
  const int tid = threadIdx.x;
  const int wave = tid >> 5, lane = tid & 31;
  const int l16 = lane & 15, hi = (lane >> 4) & 1;

  v8f acc[4];
#pragma unroll
  for (int t = 0; t < 4; ++t)
#pragma unroll
    for (int r = 0; r < 8; ++r) acc[t][r] = 0.0f;

  auto issue_A = [&](int k0, int p) {   // wave 0 only: one DMA for the whole tile
    tdm_load_2d(A + (size_t)m0 * K + k0,
                (unsigned)(uintptr_t)&As[p][0][0],
                /*tile_w=*/32, /*tile_h=*/128,
                /*tensor_w=*/32, /*tensor_h=*/128,
                /*row_stride=*/(unsigned)K);
  };
  auto issue_W = [&](int k0, int p) {   // all 256 lanes: 64x32 = 256 16B chunks
    int row = tid >> 2, col = (tid & 3) << 3;
    async_ld_b128(W + (size_t)(n0 + row) * K + k0 + col, &Ws[p][row][col]);
  };

  if (wave == 0) issue_A(0, 0);
  issue_W(0, 0);
  int p = 0;
  for (int k0 = 0; k0 < K; k0 += 32) {
    wait_async0();                                  // my W chunks landed
    if (wave == 0) __builtin_amdgcn_s_wait_tensorcnt(0);  // A tile landed
    __syncthreads();                                // publish to all waves
    if (k0 + 32 < K) {
      if (wave == 0) issue_A(k0 + 32, p ^ 1);
      issue_W(k0 + 32, p ^ 1);
    }

    v16bf af = load_fragA(&As[p][16 * wave + l16][0], hi);
#pragma unroll
    for (int nt = 0; nt < 4; ++nt) {
      v16bf bfrag = load_fragB(&Ws[p][nt * 16 + l16][0], hi);
      acc[nt] = wmma_bf16(af, bfrag, acc[nt]);
    }
    p ^= 1;
  }

  const int mrow = m0 + 16 * wave + (hi ? 8 : 0);  // rows mrow..mrow+7 <-> acc[.][r]
  if (mode == 0) {
#pragma unroll
    for (int nt = 0; nt < 4; ++nt) {
      int n = n0 + nt * 16 + l16;
#pragma unroll
      for (int r = 0; r < 8; ++r)
        outbf[(size_t)(mrow + r) * NTOT + n] = f2bf(acc[nt][r]);
    }
  } else if (mode == 1) {
#pragma unroll
    for (int nt = 0; nt < 4; ++nt) {
      int n = n0 + nt * 16 + l16;
      int h = n >> 6, d = n & 63;
      int bb = mrow / Sdim, s = mrow % Sdim;     // tile never crosses batch
      v8bf pk;
#pragma unroll
      for (int r = 0; r < 8; ++r)
        pk[r] = __builtin_bit_cast(__bf16, f2bf(acc[nt][r]));
      *reinterpret_cast<v8bf*>(
          outbf + ((size_t)((bb * 16 + h) * 64 + d)) * Sdim + s) = pk;
    }
  } else {
#pragma unroll
    for (int nt = 0; nt < 4; ++nt) {
      int n = n0 + nt * 16 + l16;
      float bv = bias[n];
#pragma unroll
      for (int r = 0; r < 8; ++r)
        outf[(size_t)(mrow + r) * NTOT + n] = acc[nt][r] + bv;
    }
  }
}

// Sliding-window causal attention, flash style. 4 waves/block; wave owns 16
// queries of one (b,h). Keys processed in 32-wide chunks inside the window.
__global__ __launch_bounds__(128) void attn_local_kernel(
    const unsigned short* __restrict__ q,    // [B*S, 1024] bf16
    const unsigned short* __restrict__ k,    // [B*S, 1024] bf16
    const unsigned short* __restrict__ vT,   // [B*16*64, S] bf16 (per-head transposed)
    const float* __restrict__ amask,         // [B*S, 1024] f32
    unsigned short* __restrict__ ctx,        // [B*S, 1024] bf16
    int S) {
  const int b = blockIdx.z, h = blockIdx.y;
  const int tid = threadIdx.x, wave = tid >> 5, lane = tid & 31;
  const int l16 = lane & 15, hi = (lane >> 4) & 1, half = hi ? 8 : 0;
  const int qw = blockIdx.x * 64 + wave * 16;  // first query row of this wave
  __shared__ unsigned short Pbuf[4][16][32];   // per-wave P tile (C->A relayout)

  const unsigned short* qrow = q + (size_t)(b * S + qw + l16) * HDIM + h * HD;
  v16bf qa0 = load_fragA(qrow, hi);        // hd 0..31
  v16bf qa1 = load_fragA(qrow + 32, hi);   // hd 32..63

  float mrow[8], lrow[8];
  v8f acc[4];
#pragma unroll
  for (int r = 0; r < 8; ++r) { mrow[r] = -1e30f; lrow[r] = 0.0f; }
#pragma unroll
  for (int t = 0; t < 4; ++t)
#pragma unroll
    for (int r = 0; r < 8; ++r) acc[t][r] = 0.0f;

  int jstart = qw - (WIN - 1);
  if (jstart < 0) jstart = 0;
  jstart &= ~31;                     // 32-aligned => all frag loads stay in [0,S)
  const int jend = qw + 15;          // last possibly-valid key (causal)

  for (int j0 = jstart; j0 <= jend; j0 += 32) {
    const unsigned short* krow0 = k + (size_t)(b * S + j0 + l16) * HDIM + h * HD;
    const unsigned short* krow1 = krow0 + (size_t)16 * HDIM;
    v8f s0, s1;
#pragma unroll
    for (int r = 0; r < 8; ++r) { s0[r] = 0.0f; s1[r] = 0.0f; }
    s0 = wmma_bf16(qa0, load_fragB(krow0, hi), s0);
    s0 = wmma_bf16(qa1, load_fragB(krow0 + 32, hi), s0);
    s1 = wmma_bf16(qa0, load_fragB(krow1, hi), s1);
    s1 = wmma_bf16(qa1, load_fragB(krow1 + 32, hi), s1);

    const int jA = j0 + l16, jB = j0 + 16 + l16;
#pragma unroll
    for (int r = 0; r < 8; ++r) {
      const int i = qw + half + r;
      float v0 = ((jA <= i) && (jA > i - WIN)) ? s0[r] : -1e30f;
      float v1 = ((jB <= i) && (jB > i - WIN)) ? s1[r] : -1e30f;
      float mx = fmaxf(v0, v1);
#pragma unroll
      for (int d = 1; d < 16; d <<= 1) mx = fmaxf(mx, __shfl_xor(mx, d, 16));
      float nm = fmaxf(mrow[r], mx);
      float sc = __expf(mrow[r] - nm);
      mrow[r] = nm;
      float p0 = (v0 > -1e29f) ? __expf(v0 - nm) : 0.0f;
      float p1 = (v1 > -1e29f) ? __expf(v1 - nm) : 0.0f;
      float ps = p0 + p1;
#pragma unroll
      for (int d = 1; d < 16; d <<= 1) ps += __shfl_xor(ps, d, 16);
      lrow[r] = lrow[r] * sc + ps;
      acc[0][r] *= sc; acc[1][r] *= sc; acc[2][r] *= sc; acc[3][r] *= sc;
      Pbuf[wave][half + r][l16]      = f2bf(p0);   // C-layout -> LDS
      Pbuf[wave][half + r][16 + l16] = f2bf(p1);
    }
    // Re-read as A operand (intra-wave DS ordering; no barrier needed)
    v16bf pf = load_fragA(&Pbuf[wave][l16][0], hi);
#pragma unroll
    for (int nt = 0; nt < 4; ++nt) {
      const unsigned short* vcol =
          vT + ((size_t)((b * 16 + h) * 64 + nt * 16 + l16)) * S + j0;
      acc[nt] = wmma_bf16(pf, load_fragB(vcol, hi), acc[nt]);
    }
  }

#pragma unroll
  for (int nt = 0; nt < 4; ++nt) {
#pragma unroll
    for (int r = 0; r < 8; ++r) {
      const int i = qw + half + r;
      const int col = h * HD + nt * 16 + l16;
      float o = (acc[nt][r] / lrow[r]) * amask[(size_t)(b * S + i) * HDIM + col];
      ctx[(size_t)(b * S + i) * HDIM + col] = f2bf(o);
    }
  }
}

extern "C" void kernel_launch(void* const* d_in, const int* in_sizes, int n_in,
                              void* d_out, int out_size, void* d_ws, size_t ws_size,
                              hipStream_t stream) {
  (void)in_sizes; (void)n_in; (void)out_size; (void)ws_size;
  const float* x  = (const float*)d_in[0];
  const float* xc = (const float*)d_in[1];
  const float* am = (const float*)d_in[2];
  const float* Wq = (const float*)d_in[3];
  const float* Wk = (const float*)d_in[4];
  const float* Wv = (const float*)d_in[5];
  const float* Wo = (const float*)d_in[6];
  const float* bo = (const float*)d_in[7];
  float* out = (float*)d_out;

  const int B = BSZ, S = SSEQ, H = HDIM;
  const size_t M  = (size_t)B * S;     // 8192
  const size_t nX = M * H;             // 8,388,608
  const size_t nW = (size_t)H * H;     // 1,048,576

  char* ws = (char*)d_ws;
  size_t off = 0;
  auto alloc = [&](size_t bytes) -> void* {
    void* p = ws + off;
    off += (bytes + 255) & ~(size_t)255;
    return p;
  };
  unsigned short* Xb  = (unsigned short*)alloc(nX * 2);
  unsigned short* XCb = (unsigned short*)alloc(nX * 2);
  unsigned short* Wqb = (unsigned short*)alloc(nW * 2);
  unsigned short* Wkb = (unsigned short*)alloc(nW * 2);
  unsigned short* Wvb = (unsigned short*)alloc(nW * 2);
  unsigned short* Wob = (unsigned short*)alloc(nW * 2);
  unsigned short* Qb  = (unsigned short*)alloc(nX * 2);
  unsigned short* Kb  = (unsigned short*)alloc(nX * 2);
  unsigned short* VTb = (unsigned short*)alloc(nX * 2);
  unsigned short* Cb  = (unsigned short*)alloc(nX * 2);

  const int cb = 256;
  cvt_bf16_kernel<<<(unsigned)((nX + cb - 1) / cb), cb, 0, stream>>>(x,  Xb,  (int)nX);
  cvt_bf16_kernel<<<(unsigned)((nX + cb - 1) / cb), cb, 0, stream>>>(xc, XCb, (int)nX);
  cvt_bf16_kernel<<<(unsigned)((nW + cb - 1) / cb), cb, 0, stream>>>(Wq, Wqb, (int)nW);
  cvt_bf16_kernel<<<(unsigned)((nW + cb - 1) / cb), cb, 0, stream>>>(Wk, Wkb, (int)nW);
  cvt_bf16_kernel<<<(unsigned)((nW + cb - 1) / cb), cb, 0, stream>>>(Wv, Wvb, (int)nW);
  cvt_bf16_kernel<<<(unsigned)((nW + cb - 1) / cb), cb, 0, stream>>>(Wo, Wob, (int)nW);

  dim3 ggrid(H / 64, (unsigned)(M / 128), 1);  // (16, 64)
  gemm_bf16_kernel<<<ggrid, 256, 0, stream>>>(Xb,  Wqb, Qb,  nullptr, nullptr, H, H, S, 0);
  gemm_bf16_kernel<<<ggrid, 256, 0, stream>>>(XCb, Wkb, Kb,  nullptr, nullptr, H, H, S, 0);
  gemm_bf16_kernel<<<ggrid, 256, 0, stream>>>(XCb, Wvb, VTb, nullptr, nullptr, H, H, S, 1);

  dim3 agrid(S / 64, NHEAD, B);  // (32, 16, 4)
  attn_local_kernel<<<agrid, 128, 0, stream>>>(Qb, Kb, VTb, am, Cb, S);

  gemm_bf16_kernel<<<ggrid, 256, 0, stream>>>(Cb, Wob, nullptr, out, bo, H, H, S, 2);
}
